// MixedTSTEncoderLayer_53334903881769
// MI455X (gfx1250) — matrix operand order repaired and checked by reading, not compiled, not run on hardware
//
#include <hip/hip_runtime.h>
#include <hip/hip_bf16.h>
#include <math.h>

// ---------------------------------------------------------------------------
// MixedTSTEncoderLayer on MI455X (gfx1250): bf16 WMMA GEMMs + f32 accumulate,
// LDS double-buffered tiles fed by async global->LDS copies (ASYNCcnt).
// ---------------------------------------------------------------------------

typedef __attribute__((ext_vector_type(16))) __bf16         v16bf;
typedef __attribute__((ext_vector_type(8)))  float          v8f;
typedef __attribute__((ext_vector_type(8)))  unsigned short us8;
typedef __attribute__((ext_vector_type(16))) unsigned short us16;

#define SCALE_QK 0.125f   // d_k = 64 -> 64^-0.5

__device__ __forceinline__ unsigned short f2b(float f) {
  unsigned u = __builtin_bit_cast(unsigned, f);
  u += 0x7FFFu + ((u >> 16) & 1u);             // round-to-nearest-even
  return (unsigned short)(u >> 16);
}
__device__ __forceinline__ float b2f(unsigned short h) {
  unsigned u = ((unsigned)h) << 16;
  return __builtin_bit_cast(float, u);
}
__device__ __forceinline__ float gelu_f(float x) {
  return 0.5f * x * (1.0f + erff(x * 0.70710678118654752f));  // exact (erf) gelu
}

__device__ __forceinline__ v16bf make_bf16x16(us8 lo, us8 hi) {
  us16 u;
#pragma unroll
  for (int i = 0; i < 8; ++i) { u[i] = lo[i]; u[8 + i] = hi[i]; }
  return __builtin_bit_cast(v16bf, u);
}

// 16-byte async copy: global (per-lane address) -> LDS (per-lane destination).
// The LDS byte offset is the low 32 bits of the flat pointer (gfx1250 LDS
// aperture keeps the offset in addr[31:0]).  The pointer is also passed as an
// (unused) operand so the LDS object escapes into the asm: combined with the
// "memory" clobber this stops LLVM from folding the later ds_loads to undef.
__device__ __forceinline__ void async_cp16(const unsigned short* lds_dst,
                                           const void* gptr) {
  const unsigned off = (unsigned)(unsigned long long)lds_dst;
  asm volatile("global_load_async_to_lds_b128 %0, %1, off"
               :: "v"(off), "v"(gptr), "v"(lds_dst) : "memory");
}
__device__ __forceinline__ void wait_async0() {
  asm volatile("s_wait_asynccnt 0x0" ::: "memory");
}

// A fragment, 16x32 bf16 (MxK). Lane l holds row (l&15);
// lanes 0-15 carry K=0..15, lanes 16-31 carry K=16..31.
__device__ __forceinline__ v16bf load_a_frag(const unsigned short* A, int lda,
                                             int rowBase, int k0) {
  const int l  = threadIdx.x & 31;
  const int m  = l & 15;
  const int kh = l >> 4;
  const unsigned short* p = A + (size_t)(rowBase + m) * lda + (k0 + kh * 16);
  us8 lo = *(const us8*)(p);        // K = k0 + kh*16 .. +7
  us8 hi = *(const us8*)(p + 8);    // K = k0 + kh*16 + 8 .. +15
  return make_bf16x16(lo, hi);
}

// B fragment, 32x16 bf16 (KxN) where B[k][n] = W[n][k], W row-major [N,K].
// Lane l holds column (l&15), K range (l>>4)*16..+15 -> contiguous in W's row.
__device__ __forceinline__ v16bf load_b_frag(const unsigned short* W, int ldw,
                                             int colBase, int k0) {
  const int l  = threadIdx.x & 31;
  const int n  = l & 15;
  const int kb = (l >> 4) * 16;
  const unsigned short* p = W + (size_t)(colBase + n) * ldw + (k0 + kb);
  us16 u = *(const us16*)p;
  return __builtin_bit_cast(v16bf, u);
}

// LDS tile variants (tile is [128 rows][64 elements], row stride 64).
__device__ __forceinline__ v16bf lds_a_frag(const unsigned short* T, int rowBase,
                                            int k0) {
  const int l = threadIdx.x & 31;
  const unsigned short* p = T + (rowBase + (l & 15)) * 64 + k0 + (l >> 4) * 16;
  us8 lo = *(const us8*)(p);
  us8 hi = *(const us8*)(p + 8);
  return make_bf16x16(lo, hi);
}
__device__ __forceinline__ v16bf lds_b_frag(const unsigned short* T, int colBase,
                                            int k0) {
  const int l = threadIdx.x & 31;
  const unsigned short* p = T + (colBase + (l & 15)) * 64 + k0 + (l >> 4) * 16;
  us16 u = *(const us16*)p;
  return __builtin_bit_cast(v16bf, u);
}

// Stage a 128x64 bf16 tile (rows row0.., K-slice k0..k0+63) into LDS via
// async copies: 1024 chunks of 16B, 256 threads -> 4 per thread.
__device__ __forceinline__ void stage_tile_async(const unsigned short* g, int ld,
                                                 int row0, int k0,
                                                 unsigned short* ldsBase) {
#pragma unroll
  for (int it = 0; it < 4; ++it) {
    const int c   = threadIdx.x + it * 256;    // 0..1023
    const int row = c >> 3, ch = c & 7;
    async_cp16(ldsBase + row * 64 + ch * 8,
               g + (size_t)(row0 + row) * ld + k0 + ch * 8);
  }
}

// ---------------------------------------------------------------------------
// GEMM: out[m,n] = sum_k A[m,k]*W[n,k] (+bias[n]) (+gelu)
// Block 256 threads (8 waves) computes a 128x128 tile; wave (mq, nh) = 32x64.
// K-loop in steps of 64 with LDS double buffering fed by async-to-LDS
// (16 wmma per wave between barriers).  K must be a multiple of 64.
// ---------------------------------------------------------------------------
__global__ void __launch_bounds__(256)
gemm_bf16_wmma(const unsigned short* __restrict__ A, int lda,
               const unsigned short* __restrict__ W, int ldw,
               const float* __restrict__ bias,
               float* __restrict__ outF, unsigned short* __restrict__ outB,
               int N, int K, int act) {
  __shared__ unsigned short tiles[4 * 128 * 64];   // A0 A1 B0 B1 = 64 KB
  unsigned short* AT0 = tiles;
  unsigned short* AT1 = tiles + 8192;
  unsigned short* BT0 = tiles + 16384;
  unsigned short* BT1 = tiles + 24576;
  const int wave  = threadIdx.x >> 5;
  const int lane  = threadIdx.x & 31;
  const int waveM = wave >> 1;            // 0..3 -> 32-row slice
  const int waveN = wave & 1;             // 0..1 -> 64-col slice
  const int rowBlk = blockIdx.x * 128;
  const int colBlk = blockIdx.y * 128;

  v8f acc[2][4] = {};
  const int nk = K >> 6;

  stage_tile_async(A, lda, rowBlk, 0, AT0);
  stage_tile_async(W, ldw, colBlk, 0, BT0);

  int buf = 0;
  for (int kc = 0; kc < nk; ++kc) {
    wait_async0();        // my wave's async copies for the current buffer done
    __syncthreads();      // everyone's done; also: all reads of other buf done
    if (kc + 1 < nk) {    // prefetch next K-slice into the other buffer
      stage_tile_async(A, lda, rowBlk, (kc + 1) * 64, buf ? AT0 : AT1);
      stage_tile_async(W, ldw, colBlk, (kc + 1) * 64, buf ? BT0 : BT1);
    }
    const unsigned short* tA = buf ? AT1 : AT0;
    const unsigned short* tB = buf ? BT1 : BT0;
#pragma unroll
    for (int kk = 0; kk < 64; kk += 32) {
      v16bf a0 = lds_a_frag(tA, waveM * 32, kk);
      v16bf a1 = lds_a_frag(tA, waveM * 32 + 16, kk);
#pragma unroll
      for (int ns = 0; ns < 4; ++ns) {
        v16bf b = lds_b_frag(tB, waveN * 64 + ns * 16, kk);
        acc[0][ns] = __builtin_amdgcn_wmma_f32_16x16x32_bf16(
            false, a0, false, b, (short)0, acc[0][ns], false, false);
        acc[1][ns] = __builtin_amdgcn_wmma_f32_16x16x32_bf16(
            false, a1, false, b, (short)0, acc[1][ns], false, false);
      }
    }
    buf ^= 1;
  }

  const int nlo = lane & 15, mh = lane >> 4;
#pragma unroll
  for (int ms = 0; ms < 2; ++ms) {
#pragma unroll
    for (int ns = 0; ns < 4; ++ns) {
      const int col = colBlk + waveN * 64 + ns * 16 + nlo;
      const float bsv = bias ? bias[col] : 0.0f;
#pragma unroll
      for (int r = 0; r < 8; ++r) {
        const int row = rowBlk + waveM * 32 + ms * 16 + r + mh * 8;
        float v = acc[ms][ns][r] + bsv;
        if (act) v = gelu_f(v);
        const size_t idx = (size_t)row * N + col;
        if (outF) outF[idx] = v;
        else      outB[idx] = f2b(v);
      }
    }
  }
}

// ---------------------------------------------------------------------------
// Temporal attention per (b, h): S = Q K^T * scale, softmax, O = P V.
// V tile (128x64) is async-prefetched into LDS while S is computed with WMMA.
// Writes merged = O + chan[b] as bf16 into Mout (linear [b*128+t, 512] layout).
// ---------------------------------------------------------------------------
__global__ void __launch_bounds__(256)
attn_wmma(const unsigned short* __restrict__ Qlin,
          const unsigned short* __restrict__ Klin,
          const unsigned short* __restrict__ Vlin,
          const float* __restrict__ chan,
          unsigned short* __restrict__ Mout) {
  __shared__ unsigned short smem[128 * 128 + 128 * 64];  // P (32KB) + Vt (16KB)
  unsigned short* P  = smem;                // probs, bf16, [128][128]
  unsigned short* Vt = smem + 16384;        // staged V, [128][64]
  const int bh = blockIdx.x;                // 0..8191
  const int b = bh >> 3, h = bh & 7;
  const int wave = threadIdx.x >> 5;
  const int lane = threadIdx.x & 31;
  const int rowBase = wave * 16;
  const int ld = 512;
  const unsigned short* Qh = Qlin + (size_t)b * 128 * ld + h * 64;
  const unsigned short* Kh = Klin + (size_t)b * 128 * ld + h * 64;
  const unsigned short* Vh = Vlin + (size_t)b * 128 * ld + h * 64;

  // ---- async prefetch of V into LDS (overlaps with S phase) ----
#pragma unroll
  for (int it = 0; it < 4; ++it) {
    const int c = threadIdx.x + it * 256;   // 128 rows x 8 chunks of 16B
    const int row = c >> 3, ch = c & 7;
    async_cp16(Vt + row * 64 + ch * 8, Vh + (size_t)row * ld + ch * 8);
  }

  // ---- S = Q K^T  (wave owns 16 rows x 128 cols) ----
  v8f s[8] = {};
  for (int k0 = 0; k0 < 64; k0 += 32) {
    v16bf a = load_a_frag(Qh, ld, rowBase, k0);
#pragma unroll
    for (int j = 0; j < 8; ++j) {
      v16bf bb = load_b_frag(Kh, ld, j * 16, k0);
      s[j] = __builtin_amdgcn_wmma_f32_16x16x32_bf16(
          false, a, false, bb, (short)0, s[j], false, false);
    }
  }

  // ---- softmax over 128 keys; row = rowBase + r + 8*(lane>>4) ----
  const int nlo = lane & 15, mh = lane >> 4;
#pragma unroll
  for (int r = 0; r < 8; ++r) {
    float mx = -3.0e38f;
#pragma unroll
    for (int j = 0; j < 8; ++j) {
      float t = s[j][r] * SCALE_QK;
      s[j][r] = t;
      mx = fmaxf(mx, t);
    }
#pragma unroll
    for (int off = 1; off < 16; off <<= 1)   // reduce across 16-lane half
      mx = fmaxf(mx, __shfl_xor(mx, off, 32));
    float sum = 0.0f;
#pragma unroll
    for (int j = 0; j < 8; ++j) {
      float e = __expf(s[j][r] - mx);
      s[j][r] = e;
      sum += e;
    }
#pragma unroll
    for (int off = 1; off < 16; off <<= 1)
      sum += __shfl_xor(sum, off, 32);
    const float inv = 1.0f / sum;
    const int row = rowBase + r + mh * 8;
#pragma unroll
    for (int j = 0; j < 8; ++j)
      P[row * 128 + j * 16 + nlo] = f2b(s[j][r] * inv);
  }
  wait_async0();            // V tile resident
  __syncthreads();

  // ---- O = P V  (A from own LDS rows, B gathered from LDS V tile) ----
  v8f o[4] = {};
  for (int k0 = 0; k0 < 128; k0 += 32) {
    v16bf a = load_a_frag(P, 128, rowBase, k0);
    const int kb = mh * 16;
#pragma unroll
    for (int j = 0; j < 4; ++j) {
      us16 u;                                // column gather of V from LDS
#pragma unroll
      for (int i = 0; i < 16; ++i)
        u[i] = Vt[(k0 + kb + i) * 64 + j * 16 + nlo];
      v16bf bb = __builtin_bit_cast(v16bf, u);
      o[j] = __builtin_amdgcn_wmma_f32_16x16x32_bf16(
          false, a, false, bb, (short)0, o[j], false, false);
    }
  }

  // ---- epilogue: merged = O + chan[b], write bf16 ----
  const float* chb = chan + (size_t)b * 512 + h * 64;
#pragma unroll
  for (int j = 0; j < 4; ++j) {
    const int col = j * 16 + nlo;
    const float cadd = chb[col];
#pragma unroll
    for (int r = 0; r < 8; ++r) {
      const int row = rowBase + r + mh * 8;
      Mout[((size_t)b * 128 + row) * 512 + h * 64 + col] = f2b(o[j][r] + cadd);
    }
  }
}

// ---------------------------------------------------------------------------
// Channel-compression path (tiny: ~1 GFLOP total) - plain VALU kernels.
// ---------------------------------------------------------------------------
__device__ __forceinline__ float block_max_256(float v, float* red) {
#pragma unroll
  for (int off = 1; off < 32; off <<= 1) v = fmaxf(v, __shfl_xor(v, off, 32));
  __syncthreads();
  if ((threadIdx.x & 31) == 0) red[threadIdx.x >> 5] = v;
  __syncthreads();
  float t = red[0];
#pragma unroll
  for (int i = 1; i < 8; ++i) t = fmaxf(t, red[i]);
  return t;
}
__device__ __forceinline__ float block_sum_256(float v, float* red) {
#pragma unroll
  for (int off = 1; off < 32; off <<= 1) v += __shfl_xor(v, off, 32);
  __syncthreads();
  if ((threadIdx.x & 31) == 0) red[threadIdx.x >> 5] = v;
  __syncthreads();
  float t = 0.0f;
#pragma unroll
  for (int i = 0; i < 8; ++i) t += red[i];
  return t;
}

// p[bs,h,q,j] = softmax_j( scale * sum_d router[h,q,d] * k[bs*32+c, h, t, d] ),
// j = c*128 + t (channel-major flatten, per reference).
__global__ void __launch_bounds__(256)
chan_scores(const float* __restrict__ router, const unsigned short* __restrict__ Klin,
            float* __restrict__ p) {
  __shared__ float rr[64];
  __shared__ float red[8];
  const int bhq = blockIdx.x;                 // bs*64 + h*8 + q
  const int q = bhq & 7, h = (bhq >> 3) & 7, bs = bhq >> 6;
  if (threadIdx.x < 64)
    rr[threadIdx.x] = router[((size_t)h * 8 + q) * 64 + threadIdx.x];
  __syncthreads();

  float sv[16];
  float mx = -3.0e38f;
#pragma unroll
  for (int it = 0; it < 16; ++it) {
    const int j = threadIdx.x + it * 256;     // 0..4095
    const int c = j >> 7, t = j & 127;
    const unsigned short* kp =
        Klin + ((size_t)(bs * 32 + c) * 128 + t) * 512 + h * 64;
    float acc = 0.0f;
    for (int d = 0; d < 64; ++d) acc += rr[d] * b2f(kp[d]);
    sv[it] = acc * SCALE_QK;
    mx = fmaxf(mx, sv[it]);
  }
  mx = block_max_256(mx, red);
  float sum = 0.0f;
#pragma unroll
  for (int it = 0; it < 16; ++it) { sv[it] = __expf(sv[it] - mx); sum += sv[it]; }
  sum = block_sum_256(sum, red);
  const float inv = 1.0f / sum;
#pragma unroll
  for (int it = 0; it < 16; ++it)
    p[(size_t)bhq * 4096 + threadIdx.x + it * 256] = sv[it] * inv;
}

// oc_t[bs, q, h*64+d] = sum_j p[bs,h,q,j] * v[bs*32 + j%32, h, j/32, d] + pos[q, h*64+d]
// (value flatten is time-major: vc row j -> (t=j/32, c=j%32), per reference quirk.)
__global__ void __launch_bounds__(64)
chan_oc(const float* __restrict__ p, const unsigned short* __restrict__ Vlin,
        const float* __restrict__ pos, float* __restrict__ ocb) {
  const int bhq = blockIdx.x;
  const int q = bhq & 7, h = (bhq >> 3) & 7, bs = bhq >> 6;
  const int d = threadIdx.x;                  // 0..63
  const float* pp = p + (size_t)bhq * 4096;
  float acc = 0.0f;
  for (int j = 0; j < 4096; ++j) {
    const int c = j & 31, t = j >> 5;
    acc += pp[j] * b2f(Vlin[((size_t)(bs * 32 + c) * 128 + t) * 512 + h * 64 + d]);
  }
  const int a = h * 64 + d;
  ocb[((size_t)bs * 8 + q) * 512 + a] = acc + pos[(size_t)q * 512 + a];
}

// chan[bs*32+c, a] = sum_q m_expand[c,q] * oc_t[bs,q,a]
__global__ void __launch_bounds__(256)
chan_expand(const float* __restrict__ ocb, const float* __restrict__ mexp,
            float* __restrict__ chan) {
  const int row = blockIdx.x;                 // b = bs*32 + c
  const int bs = row >> 5, c = row & 31;
  for (int a = threadIdx.x; a < 512; a += 256) {
    float acc = 0.0f;
#pragma unroll
    for (int q = 0; q < 8; ++q)
      acc += mexp[(size_t)c * 8 + q] * ocb[((size_t)bs * 8 + q) * 512 + a];
    chan[(size_t)row * 512 + a] = acc;
  }
}

// ---------------------------------------------------------------------------
// LayerNorm over 512 with fused residual: out = LN(a+b)*g + bias.
// ---------------------------------------------------------------------------
__global__ void __launch_bounds__(256)
ln_kernel(const float* __restrict__ a, const float* __restrict__ b,
          const float* __restrict__ g, const float* __restrict__ bias,
          float* __restrict__ outF, unsigned short* __restrict__ outB) {
  __shared__ float red[8];
  const size_t row = blockIdx.x;
  const int t = threadIdx.x;
  const float* pa = a + row * 512;
  const float* pb = b + row * 512;
  const float v0 = pa[t] + pb[t];
  const float v1 = pa[t + 256] + pb[t + 256];
  const float mean = block_sum_256(v0 + v1, red) * (1.0f / 512.0f);
  const float d0 = v0 - mean, d1 = v1 - mean;
  const float var = block_sum_256(d0 * d0 + d1 * d1, red) * (1.0f / 512.0f);
  const float inv = rsqrtf(var + 1e-5f);
  const float o0 = d0 * inv * g[t] + bias[t];
  const float o1 = d1 * inv * g[t + 256] + bias[t + 256];
  if (outF) { outF[row * 512 + t] = o0; outF[row * 512 + t + 256] = o1; }
  if (outB) { outB[row * 512 + t] = f2b(o0); outB[row * 512 + t + 256] = f2b(o1); }
}

__global__ void __launch_bounds__(256)
cvt_f32_bf16(const float* __restrict__ in, unsigned short* __restrict__ out, int n) {
  const int i = blockIdx.x * 256 + threadIdx.x;
  if (i < n) out[i] = f2b(in[i]);
}

// ---------------------------------------------------------------------------
// Launcher
// ---------------------------------------------------------------------------
extern "C" void kernel_launch(void* const* d_in, const int* in_sizes, int n_in,
                              void* d_out, int out_size, void* d_ws, size_t ws_size,
                              hipStream_t stream) {
  (void)in_sizes; (void)n_in; (void)out_size; (void)ws_size;
  const float* src    = (const float*)d_in[0];
  const float* Wq     = (const float*)d_in[1];
  const float* bq     = (const float*)d_in[2];
  const float* Wk     = (const float*)d_in[3];
  const float* bk     = (const float*)d_in[4];
  const float* Wv     = (const float*)d_in[5];
  const float* bv     = (const float*)d_in[6];
  const float* Wo     = (const float*)d_in[7];
  const float* bo     = (const float*)d_in[8];
  const float* router = (const float*)d_in[9];
  const float* mexp   = (const float*)d_in[10];
  const float* pos    = (const float*)d_in[11];
  const float* g1     = (const float*)d_in[12];
  const float* b1     = (const float*)d_in[13];
  const float* Wf1    = (const float*)d_in[14];
  const float* bf1    = (const float*)d_in[15];
  const float* Wf2    = (const float*)d_in[16];
  const float* bf2    = (const float*)d_in[17];
  const float* g2     = (const float*)d_in[18];
  const float* b2     = (const float*)d_in[19];
  float* out = (float*)d_out;

  // Workspace layout (bytes). M = 1024*128 = 131072 rows.
  char* ws = (char*)d_ws;
  unsigned short* sB    = (unsigned short*)(ws + 0ull);             // 128 MiB src bf16 (reused: merged)
  unsigned short* qB    = (unsigned short*)(ws + 134217728ull);     // 128 MiB (reused: x1 bf16)
  unsigned short* kB    = (unsigned short*)(ws + 268435456ull);     // 128 MiB
  unsigned short* vB    = (unsigned short*)(ws + 402653184ull);     // 128 MiB
  float*          yF    = (float*)(ws + 536870912ull);              // 256 MiB attn_out f32 (reused: ff2)
  float*          x1F   = (float*)(ws + 805306368ull);              // 256 MiB
  unsigned short* h1B   = (unsigned short*)(ws + 1073741824ull);    // 512 MiB gelu hidden bf16
  unsigned short* wqB   = (unsigned short*)(ws + 1610612736ull);
  unsigned short* wkB   = (unsigned short*)(ws + 1611137024ull);
  unsigned short* wvB   = (unsigned short*)(ws + 1611661312ull);
  unsigned short* woB   = (unsigned short*)(ws + 1612185600ull);
  unsigned short* wf1B  = (unsigned short*)(ws + 1612709888ull);
  unsigned short* wf2B  = (unsigned short*)(ws + 1614807040ull);
  float*          pC    = (float*)(ws + 1616904192ull);             // 4 MiB softmax probs
  float*          ocB   = (float*)(ws + 1621098496ull);             // 512 KiB oc_t
  float*          chanB = (float*)(ws + 1621622784ull);             // 2 MiB chan
  unsigned short* mB  = sB;   // merged (out_tf + chan) bf16
  unsigned short* x1B = qB;   // LN1 output bf16
  float*          f2F = yF;   // FFN2 output f32

  // fp32 -> bf16
  cvt_f32_bf16<<<262144, 256, 0, stream>>>(src, sB, 67108864);
  cvt_f32_bf16<<<1024, 256, 0, stream>>>(Wq, wqB, 262144);
  cvt_f32_bf16<<<1024, 256, 0, stream>>>(Wk, wkB, 262144);
  cvt_f32_bf16<<<1024, 256, 0, stream>>>(Wv, wvB, 262144);
  cvt_f32_bf16<<<1024, 256, 0, stream>>>(Wo, woB, 262144);
  cvt_f32_bf16<<<4096, 256, 0, stream>>>(Wf1, wf1B, 1048576);
  cvt_f32_bf16<<<4096, 256, 0, stream>>>(Wf2, wf2B, 1048576);

  const dim3 blk(256);
  // Q/K/V projections: [131072,512] x [512,512]^T  (128x128 block tiles)
  gemm_bf16_wmma<<<dim3(1024, 4), blk, 0, stream>>>(sB, 512, wqB, 512, bq, nullptr, qB, 512, 512, 0);
  gemm_bf16_wmma<<<dim3(1024, 4), blk, 0, stream>>>(sB, 512, wkB, 512, bk, nullptr, kB, 512, 512, 0);
  gemm_bf16_wmma<<<dim3(1024, 4), blk, 0, stream>>>(sB, 512, wvB, 512, bv, nullptr, vB, 512, 512, 0);
  // channel compression path
  chan_scores<<<2048, 256, 0, stream>>>(router, kB, pC);
  chan_oc<<<2048, 64, 0, stream>>>(pC, vB, pos, ocB);
  chan_expand<<<1024, 256, 0, stream>>>(ocB, mexp, chanB);
  // temporal attention + merge with chan
  attn_wmma<<<8192, 256, 0, stream>>>(qB, kB, vB, chanB, mB);
  // output projection -> f32
  gemm_bf16_wmma<<<dim3(1024, 4), blk, 0, stream>>>(mB, 512, woB, 512, bo, yF, nullptr, 512, 512, 0);
  // LN1(src + attn_out) -> x1 (f32 + bf16)
  ln_kernel<<<131072, 256, 0, stream>>>(src, yF, g1, b1, x1F, x1B);
  // FFN: gelu(x1 Wf1^T + bf1) -> bf16; then Wf2^T + bf2 -> f32
  gemm_bf16_wmma<<<dim3(1024, 16), blk, 0, stream>>>(x1B, 512, wf1B, 512, bf1, nullptr, h1B, 2048, 512, 1);
  gemm_bf16_wmma<<<dim3(1024, 4), blk, 0, stream>>>(h1B, 2048, wf2B, 2048, bf2, f2F, nullptr, 512, 2048, 0);
  // LN2(x1 + ff) -> out
  ln_kernel<<<131072, 256, 0, stream>>>(x1F, f2F, g2, b2, out, nullptr);
}
